// PerResidueEncoderV2_455266533563
// MI455X (gfx1250) — compile-verified, matrix operand order checked by reading, not run on hardware
//
#include <hip/hip_runtime.h>
#include <hip/hip_bf16.h>
#include <math.h>

// Problem constants (from reference)
#define NB      16
#define LRES    2048
#define NROWS   (NB*LRES)        // 32768
// W1 row blocks: [0,128) aa_feat, [128,1118) crd (aa*45+atom*3+i), [1118,1157) dihed, [1157,1285) type

typedef __attribute__((ext_vector_type(16))) _Float16 v16h;
typedef __attribute__((ext_vector_type(8)))  float    v8f;

struct F3 { float x, y, z; };
__device__ __forceinline__ F3 f3sub(F3 a, F3 b){ return {a.x-b.x, a.y-b.y, a.z-b.z}; }
__device__ __forceinline__ F3 f3cross(F3 a, F3 b){
  return {a.y*b.z-a.z*b.y, a.z*b.x-a.x*b.z, a.x*b.y-a.y*b.x};
}
__device__ __forceinline__ float f3dot(F3 a, F3 b){ return a.x*b.x + a.y*b.y + a.z*b.z; }

__device__ __forceinline__ float dihedral(F3 p0, F3 p1, F3 p2, F3 p3){
  F3 v0 = f3sub(p2,p1), v1 = f3sub(p0,p1), v2 = f3sub(p3,p2);
  F3 u1 = f3cross(v0,v1), u2 = f3cross(v0,v2);
  float n1 = sqrtf(f3dot(u1,u1)), n2 = sqrtf(f3dot(u2,u2));
  if (!(n1 > 0.f) || !(n2 > 0.f)) return 0.f;   // matches nan_to_num(NaN)->0
  float c = f3dot(u1,u2) / (n1*n2);
  c = fminf(fmaxf(c, -0.999999f), 0.999999f);
  float s = f3dot(f3cross(v1,v2), v0);
  float sg = (s > 0.f) ? 1.f : ((s < 0.f) ? -1.f : 0.f);
  return sg * acosf(c);
}

// ---------------------------------------------------------------------------
// Fragment index maps (CDNA5 v_wmma_f32_16x16x32_f16, wave32)
// A 16x32 f16: lane = M + 16*((K>>3)&1); element j: K = (j>>3)*16 + (lane>>4)*8 + (j&7)
//   inverse: j = ((K>>4)<<3) | (K&7)
// B 32x16 f16: mirrored with N in lane low bits.
// C/D f32: element r -> (M = r + 8*(lane>>4), N = lane&15)
// Fragments are stored "lane-contiguous": frag f, lane l -> 16 f16 at (f*32+l)*16.
// ---------------------------------------------------------------------------

// ---------------------------------------------------------------------------
// Kernel 0: tiny precomputed gather tables
//   T_aa[a][n]   = sum_k aatype_W[a][k] * W1[k][n]          (22 x 256)
//   T_type[f][n] = sum_k type_W[f][k]   * W1[1157+k][n]     (10 x 256)
// ---------------------------------------------------------------------------
__global__ void k_tables(const float* __restrict__ aatype_W,
                         const float* __restrict__ type_W,
                         const float* __restrict__ W1,
                         float* __restrict__ T_aa,
                         float* __restrict__ T_type)
{
  int b = blockIdx.x, t = threadIdx.x;
  if (b < 22){
    float acc = 0.f;
    #pragma unroll 8
    for (int k = 0; k < 128; ++k) acc += aatype_W[b*128 + k] * W1[(size_t)k*256 + t];
    T_aa[b*256 + t] = acc;
  } else {
    int f = b - 22;                       // 0..9
    float acc = 0.f;
    #pragma unroll 8
    for (int k = 0; k < 128; ++k) acc += type_W[f*128 + k] * W1[(size_t)(1157 + k)*256 + t];
    T_type[f*256 + t] = acc;
  }
}

// ---------------------------------------------------------------------------
// Kernel 1: per-residue features + layer 1 (exploits one-hot sparsity of crd_feat)
//   one block per residue row; thread t computes h1[row][t]
//   H1 is written directly in A-fragment order for the WMMA kernel:
//     frag (rt = row>>4, kk = t>>5) at ((rt*8+kk)*32 + lane)*16 + j
// ---------------------------------------------------------------------------
__global__ __launch_bounds__(256)
void k_encode(const int*  __restrict__ aa,
              const float* __restrict__ pos,
              const int*  __restrict__ chain_nb,
              const int*  __restrict__ res_nb,
              const unsigned char* __restrict__ mask_atoms,
              const int*  __restrict__ frag,
              const float* __restrict__ R,
              const float* __restrict__ t,
              const float* __restrict__ W1,
              const float* __restrict__ b1,
              const float* __restrict__ T_aa,
              const float* __restrict__ T_type,
              _Float16* __restrict__ H1)
{
  __shared__ float s_crd[45];
  __shared__ float s_dih[39];
  __shared__ float s_ang[3];
  __shared__ float s_am[3];
  __shared__ int   s_aa, s_fr;

  const int row = blockIdx.x;
  const int nb  = row >> 11;
  const int l   = row & (LRES - 1);
  const int tid = threadIdx.x;

  if (tid < 45){
    // crd[atom][i] = sum_j R[j][i] * (pos[atom][j] - t[j]), masked
    int atom = tid / 3, i = tid - atom*3;
    size_t pb = ((size_t)row*15 + atom)*3;
    float vx = pos[pb+0] - t[(size_t)row*3+0];
    float vy = pos[pb+1] - t[(size_t)row*3+1];
    float vz = pos[pb+2] - t[(size_t)row*3+2];
    const float* Rr = R + (size_t)row*9;
    float c = Rr[0*3+i]*vx + Rr[1*3+i]*vy + Rr[2*3+i]*vz;
    float m = mask_atoms[(size_t)row*15 + atom] ? 1.f : 0.f;
    s_crd[tid] = c * m;
  } else if (tid < 48){
    int which = tid - 45;                 // 0=omega, 1=phi, 2=psi
    bool valid = false; float ang = 0.f;

    auto mres = [&](int ll)->bool { return mask_atoms[((size_t)(nb*LRES+ll))*15 + 1] != 0; };
    auto consec = [&](int la)->bool {     // consec(la, la+1)
      int lb = la + 1;
      return (res_nb[nb*LRES+lb] - res_nb[nb*LRES+la] == 1) &&
             (chain_nb[nb*LRES+lb] == chain_nb[nb*LRES+la]) &&
             mres(la) && mres(lb);
    };
    auto atomp = [&](int ll, int a)->F3 {
      const float* p = pos + (((size_t)(nb*LRES+ll))*15 + a)*3;
      return F3{p[0], p[1], p[2]};
    };
    // atoms: 0=N, 1=CA, 2=C
    if (which == 0){        // omega: dihed(CA[l-1], C[l-1], N[l], CA[l]), mask ~n_term
      if (l >= 1 && consec(l-1)){ valid = true;
        ang = dihedral(atomp(l-1,1), atomp(l-1,2), atomp(l,0), atomp(l,1)); }
    } else if (which == 1){ // phi: dihed(C[l-1], N[l], CA[l], C[l])
      if (l >= 1 && consec(l-1)){ valid = true;
        ang = dihedral(atomp(l-1,2), atomp(l,0), atomp(l,1), atomp(l,2)); }
    } else {                // psi: dihed(N[l], CA[l], C[l], N[l+1]), mask ~c_term
      if (l <= LRES-2 && consec(l)){ valid = true;
        ang = dihedral(atomp(l,0), atomp(l,1), atomp(l,2), atomp(l+1,0)); }
    }
    s_ang[which] = ang;
    s_am[which]  = valid ? 1.f : 0.f;
  } else if (tid == 48){ s_aa = aa[row]; }
  else if (tid == 49){ s_fr = frag[row]; }
  __syncthreads();

  if (tid < 39){
    // angular encoding: per angle 13 = [x, sin(x*f), cos(x*f)], f = {1,2,3,1,1/2,1/3}
    int ai = tid / 13, j = tid - ai*13;
    const float F[6] = {1.f, 2.f, 3.f, 1.f, 0.5f, 1.f/3.f};
    float x = s_ang[ai], m = s_am[ai];
    float v;
    if (j == 0)      v = x;
    else if (j < 7)  v = sinf(x * F[j-1]);
    else             v = cosf(x * F[j-7]);
    s_dih[tid] = m * v;                   // mask pre- and post-encoding collapses to this
  }
  __syncthreads();

  // ---- layer 1: h1[t] = relu( T_aa[aa][t] + T_type[fr][t] + b1[t]
  //                             + crd . W1_crd(aa)[:,t] + dihed . W1_dihed[:,t] )
  {
    int a = s_aa, f = s_fr;
    float acc = T_aa[a*256 + tid] + T_type[f*256 + tid] + b1[tid];
    const float* Wc = W1 + ((size_t)(128 + a*45))*256 + tid;
    #pragma unroll 5
    for (int k = 0; k < 45; ++k) acc += s_crd[k] * Wc[(size_t)k*256];
    const float* Wd = W1 + (size_t)1118*256 + tid;
    #pragma unroll
    for (int j = 0; j < 39; ++j) acc += s_dih[j] * Wd[(size_t)j*256];

    // store in A-fragment order
    int m  = row & 15;
    int rt = row >> 4;
    int kk = tid >> 5;
    int k  = tid & 31;
    int j  = ((k >> 4) << 3) | (k & 7);
    int ln = m + (((k >> 3) & 1) << 4);
    H1[(((size_t)rt*8 + kk)*32 + ln)*16 + j] = (_Float16)fmaxf(acc, 0.f);
  }
}

// ---------------------------------------------------------------------------
// WMMA helpers
// ---------------------------------------------------------------------------
__device__ __forceinline__ v8f wmma_f16(v16h a, v16h b, v8f c){
  return __builtin_amdgcn_wmma_f32_16x16x32_f16(false, a, false, b, (short)0, c, false, false);
}

// Stage one 32x16 B-fragment of a (Kx128 row-major f32) weight into LDS,
// lane-contiguous (32 B per lane -> v16h load = 2x ds_load_b128 at use site).
__device__ __forceinline__ void stage_b_frag(const float* __restrict__ W,
                                             _Float16* dst, int f, int lane)
{
  int kt = f >> 3, nt = f & 7;
  int n  = nt*16 + (lane & 15);
  int kb = kt*32 + ((lane >> 4) << 3);
  v16h v;
  #pragma unroll
  for (int j = 0; j < 16; ++j){
    int k = kb + ((j >> 3) << 4) + (j & 7);
    v[j] = (_Float16)W[(size_t)k*128 + n];
  }
  *(v16h*)(dst + ((size_t)f*32 + lane)*16) = v;
}

// ---------------------------------------------------------------------------
// Kernel 2: fused layers 2-4, f16 WMMA
//   block = 256 threads = 8 waves; each wave does two 16-row tiles (256 rows/block)
//   All operands pre-swizzled into fragment order: A loads are contiguous v16h,
//   B loads are contiguous v16h from LDS (ds_load_b128 pairs).
//   k-tile outer / n-tile inner with 8 independent accumulators (no serial D->C chain).
//   LDS: W2f 64KB | W3f 32KB | W4f 32KB | sHa 32KB | sHb 32KB = 192KB dynamic
// ---------------------------------------------------------------------------
__global__ __launch_bounds__(256)
void k_mlp(const _Float16* __restrict__ H1,
           const float* __restrict__ W2, const float* __restrict__ b2,
           const float* __restrict__ W3, const float* __restrict__ b3,
           const float* __restrict__ W4, const float* __restrict__ b4,
           const unsigned char* __restrict__ mask_atoms,
           float* __restrict__ out)
{
  extern __shared__ __attribute__((aligned(32))) _Float16 smem[];
  _Float16* sW2f = smem;                   // 64 frags * 32 lanes * 16 = 32768 h
  _Float16* sW3f = sW2f + 64*32*16;        // 32 frags = 16384 h
  _Float16* sW4f = sW3f + 32*32*16;        // 32 frags = 16384 h
  _Float16* sHa  = sW4f + 32*32*16;        // 8 waves * 4 frags * 32*16 = 16384 h
  _Float16* sHb  = sHa  + 8*4*32*16;       // 16384 h

  const int lane = threadIdx.x & 31;
  const int wave = threadIdx.x >> 5;
  const int n    = lane & 15;
  const int half = lane >> 4;

  // ---- stage weights into B-fragment order (round-robin frags over waves)
  for (int f = wave; f < 64; f += 8) stage_b_frag(W2, sW2f, f, lane);
  for (int f = wave; f < 32; f += 8) stage_b_frag(W3, sW3f, f, lane);
  for (int f = wave; f < 32; f += 8) stage_b_frag(W4, sW4f, f, lane);
  __syncthreads();

  _Float16* myA = sHa + wave*(4*32*16);
  _Float16* myB = sHb + wave*(4*32*16);

  for (int p = 0; p < 2; ++p){
    const int rt   = blockIdx.x*16 + wave*2 + p;   // global 16-row tile index
    const int row0 = rt*16;

    // per-lane residue mask for the 8 accumulator rows (mask_residue = mask_atoms[:,:,CA=1])
    float msk[8];
    #pragma unroll
    for (int r = 0; r < 8; ++r){
      int rid = row0 + r + 8*half;
      msk[r] = mask_atoms[(size_t)rid*15 + 1] ? 1.f : 0.f;
    }

    // ---- layer 2: h2 = relu(H1 @ W2 + b2)   (K=256, N=128)
    v16h a1[8];
    #pragma unroll
    for (int kk = 0; kk < 8; ++kk)
      a1[kk] = *(const v16h*)(H1 + (((size_t)rt*8 + kk)*32 + lane)*16);

    {
      v8f acc[8];
      #pragma unroll
      for (int nt = 0; nt < 8; ++nt){
        float bias = b2[nt*16 + n];
        #pragma unroll
        for (int r = 0; r < 8; ++r) acc[nt][r] = bias;
      }
      #pragma unroll
      for (int kk = 0; kk < 8; ++kk){
        #pragma unroll
        for (int nt = 0; nt < 8; ++nt){
          v16h b = *(const v16h*)(sW2f + (((kk << 3) | nt)*32 + lane)*16);
          acc[nt] = wmma_f16(a1[kk], b, acc[nt]);
        }
      }
      // relu + store to myA in A-fragment order for next layer
      #pragma unroll
      for (int nt = 0; nt < 8; ++nt){
        #pragma unroll
        for (int r = 0; r < 8; ++r){
          int m   = r + 8*half;
          int dst = (((nt >> 1)*32) + m + ((n >> 3) << 4))*16 + ((nt & 1) << 3) + (n & 7);
          myA[dst] = (_Float16)fmaxf(acc[nt][r], 0.f);
        }
      }
    }

    // ---- layer 3: h3 = relu(h2 @ W3 + b3)   (K=128, N=128)
    v16h a2[4];
    #pragma unroll
    for (int kk = 0; kk < 4; ++kk)
      a2[kk] = *(const v16h*)(myA + ((size_t)kk*32 + lane)*16);

    {
      v8f acc[8];
      #pragma unroll
      for (int nt = 0; nt < 8; ++nt){
        float bias = b3[nt*16 + n];
        #pragma unroll
        for (int r = 0; r < 8; ++r) acc[nt][r] = bias;
      }
      #pragma unroll
      for (int kk = 0; kk < 4; ++kk){
        #pragma unroll
        for (int nt = 0; nt < 8; ++nt){
          v16h b = *(const v16h*)(sW3f + (((kk << 3) | nt)*32 + lane)*16);
          acc[nt] = wmma_f16(a2[kk], b, acc[nt]);
        }
      }
      #pragma unroll
      for (int nt = 0; nt < 8; ++nt){
        #pragma unroll
        for (int r = 0; r < 8; ++r){
          int m   = r + 8*half;
          int dst = (((nt >> 1)*32) + m + ((n >> 3) << 4))*16 + ((nt & 1) << 3) + (n & 7);
          myB[dst] = (_Float16)fmaxf(acc[nt][r], 0.f);
        }
      }
    }

    // ---- layer 4: out = (h3 @ W4 + b4) * mask_residue   (K=128, N=128) -> f32
    v16h a3[4];
    #pragma unroll
    for (int kk = 0; kk < 4; ++kk)
      a3[kk] = *(const v16h*)(myB + ((size_t)kk*32 + lane)*16);

    {
      v8f acc[8];
      #pragma unroll
      for (int nt = 0; nt < 8; ++nt){
        float bias = b4[nt*16 + n];
        #pragma unroll
        for (int r = 0; r < 8; ++r) acc[nt][r] = bias;
      }
      #pragma unroll
      for (int kk = 0; kk < 4; ++kk){
        #pragma unroll
        for (int nt = 0; nt < 8; ++nt){
          v16h b = *(const v16h*)(sW4f + (((kk << 3) | nt)*32 + lane)*16);
          acc[nt] = wmma_f16(a3[kk], b, acc[nt]);
        }
      }
      #pragma unroll
      for (int nt = 0; nt < 8; ++nt){
        #pragma unroll
        for (int r = 0; r < 8; ++r){
          size_t rid = (size_t)(row0 + r + 8*half);
          out[rid*128 + nt*16 + n] = acc[nt][r] * msk[r];
        }
      }
    }
  }
}

// ---------------------------------------------------------------------------
// Host launcher
// ---------------------------------------------------------------------------
extern "C" void kernel_launch(void* const* d_in, const int* in_sizes, int n_in,
                              void* d_out, int out_size, void* d_ws, size_t ws_size,
                              hipStream_t stream)
{
  (void)in_sizes; (void)n_in; (void)out_size; (void)ws_size;

  const int*   aa        = (const int*)  d_in[0];
  const float* pos       = (const float*)d_in[1];
  const int*   chain_nb  = (const int*)  d_in[2];
  const int*   res_nb    = (const int*)  d_in[3];
  const unsigned char* mask_atoms = (const unsigned char*)d_in[4];   // bool array
  const int*   frag      = (const int*)  d_in[5];
  const float* R         = (const float*)d_in[6];
  const float* t         = (const float*)d_in[7];
  const float* aatype_W  = (const float*)d_in[8];
  const float* type_W    = (const float*)d_in[9];
  const float* W1        = (const float*)d_in[10];
  const float* b1        = (const float*)d_in[11];
  const float* W2        = (const float*)d_in[12];
  const float* b2        = (const float*)d_in[13];
  const float* W3        = (const float*)d_in[14];
  const float* b3        = (const float*)d_in[15];
  const float* W4        = (const float*)d_in[16];
  const float* b4        = (const float*)d_in[17];
  float* out             = (float*)d_out;

  // workspace layout
  float*    T_aa   = (float*)d_ws;                              // 22*256 f32
  float*    T_type = (float*)((char*)d_ws + 32768);             // 10*256 f32
  _Float16* H1     = (_Float16*)((char*)d_ws + 65536);          // 32768*256 f16 (16 MB), A-frag order

  k_tables<<<32, 256, 0, stream>>>(aatype_W, type_W, W1, T_aa, T_type);

  k_encode<<<NROWS, 256, 0, stream>>>(aa, pos, chain_nb, res_nb, mask_atoms, frag,
                                      R, t, W1, b1, T_aa, T_type, H1);

  // LDS: (64 + 32 + 32) weight frags + 2 * 8-wave activation frag buffers
  const size_t lds_bytes = (size_t)(64*32*16 + 32*32*16 + 32*32*16 + 2*8*4*32*16) * sizeof(_Float16); // 192 KB
  k_mlp<<<NROWS/256, 256, lds_bytes, stream>>>(H1, W2, b2, W3, b3, W4, b4, mask_atoms, out);
}
